// TargetPred_79697413145183
// MI455X (gfx1250) — matrix operand order repaired and checked by reading, not compiled, not run on hardware
//
#include <hip/hip_runtime.h>
#include <hip/hip_bf16.h>
#include <math.h>

// Problem constants (match reference)
#define B_    256
#define N_    2048
#define C_    128
#define DIN_  130
#define H_    64
#define M_    50
#define K1_   160            // DIN padded to multiple of 32 (5 k-steps)
#define K1P_  168            // LDS row stride (halves), keeps 16B-aligned chunks
#define K2P_  72             // LDS row stride for K=64 stages
#define ROWS_ 128            // rows per block (4 waves x 32 rows)

typedef __attribute__((ext_vector_type(16))) _Float16 v16h;
typedef __attribute__((ext_vector_type(8)))  float    v8f;
typedef __attribute__((ext_vector_type(4)))  unsigned int u32x4;

union FragU { u32x4 q[2]; v16h h; };

// A fragment, f16 16x32 (MxK). Lanes 0-15: M=lane, K=kb..kb+7 (VGPR0-3),
// K=kb+16..kb+23 (VGPR4-7) with kb=k0; lanes 16-31 same with kb=k0+8.
__device__ __forceinline__ v16h load_frag_a(const _Float16* xs, int stride, int k0, int lane) {
  const int row = lane & 15;
  const int kb  = k0 + ((lane & 16) ? 8 : 0);
  const _Float16* p = xs + row * stride + kb;
  FragU f;
  f.q[0] = *(const u32x4*)(p);        // K = kb .. kb+7
  f.q[1] = *(const u32x4*)(p + 16);   // K = kb+16 .. kb+23
  return f.h;
}

// B fragment, f16 32x16 (KxN), from K-major (transposed) weight image.
// Lanes 0-15: N=lane, K=kb..kb+7, kb+16..kb+23 with kb=k0; lanes 16-31 same with kb=k0+8.
__device__ __forceinline__ v16h load_frag_b(const _Float16* wt, int stride, int c0, int k0, int lane) {
  const int col = c0 + (lane & 15);
  const int kb  = k0 + ((lane & 16) ? 8 : 0);
  const _Float16* p = wt + col * stride + kb;
  FragU f;
  f.q[0] = *(const u32x4*)(p);
  f.q[1] = *(const u32x4*)(p + 16);
  return f.h;
}

__device__ __forceinline__ v8f wmma_f16(v16h a, v16h b, v8f c) {
  return __builtin_amdgcn_wmma_f32_16x16x32_f16(false, a, false, b, (short)0, c, false, false);
}

__device__ __forceinline__ v8f zero8() {
  v8f z = {0.f,0.f,0.f,0.f,0.f,0.f,0.f,0.f};
  return z;
}

// ---------------------------------------------------------------------------
// Kernel 1: fused 3-layer MLP (both heads) over a 128-row tile of one batch.
// block = 128 threads (4 wave32), each wave owns 32 rows (2 A-tiles sharing
// every B fragment). grid = (N/128, B).
// ---------------------------------------------------------------------------
__global__ __launch_bounds__(128) void mlp_kernel(
    const float* __restrict__ feat, const float* __restrict__ cand, const float* __restrict__ mask,
    const float* __restrict__ pw1, const float* __restrict__ pb1, const float* __restrict__ pg1, const float* __restrict__ pbe1,
    const float* __restrict__ pw2, const float* __restrict__ pb2, const float* __restrict__ pg2, const float* __restrict__ pbe2,
    const float* __restrict__ pw3, const float* __restrict__ pb3,
    const float* __restrict__ ow1, const float* __restrict__ ob1, const float* __restrict__ og1, const float* __restrict__ obe1,
    const float* __restrict__ ow2, const float* __restrict__ ob2, const float* __restrict__ og2, const float* __restrict__ obe2,
    const float* __restrict__ ow3, const float* __restrict__ ob3,
    float* __restrict__ logits, float* __restrict__ offs)
{
  extern __shared__ char smem[];
  _Float16* Xs  = (_Float16*)smem;           // ROWS_ x K1P_  (input tile, f16)
  _Float16* W1t = Xs  + ROWS_ * K1P_;        // 64 x K1P_     (W1 transposed)
  _Float16* W2t = W1t + 64 * K1P_;           // 64 x K2P_
  _Float16* W3t = W2t + 64 * K2P_;           // 16 x K2P_
  _Float16* X2  = W3t + 16 * K2P_;           // ROWS_ x K2P_  (normalized acts)
  float*    Hs  = (float*)(X2 + ROWS_ * K2P_); // ROWS_ x 64  (f32 pre-LN)

  const int b    = blockIdx.y;
  const int n0   = blockIdx.x * ROWS_;
  const int tid  = threadIdx.x;
  const int wave = tid >> 5;
  const int lane = tid & 31;

  // ---- stage 0: build padded input tile [feat(128) | cand(2) | zeros] ----
  for (int idx = tid; idx < ROWS_ * K1_; idx += 128) {
    const int r = idx / K1_;
    const int k = idx - r * K1_;
    float v = 0.f;
    if (k < C_)        v = feat[b * C_ + k];
    else if (k < DIN_) v = cand[((size_t)(b * N_ + n0 + r)) * 2 + (k - C_)];
    Xs[r * K1P_ + k] = (_Float16)v;
  }
  __syncthreads();

  for (int head = 0; head < 2; ++head) {
    const float* w1  = head ? ow1  : pw1;  const float* b1  = head ? ob1  : pb1;
    const float* g1  = head ? og1  : pg1;  const float* be1 = head ? obe1 : pbe1;
    const float* w2  = head ? ow2  : pw2;  const float* b2  = head ? ob2  : pb2;
    const float* g2  = head ? og2  : pg2;  const float* be2 = head ? obe2 : pbe2;
    const float* w3  = head ? ow3  : pw3;  const float* b3  = head ? ob3  : pb3;
    const int ncol3  = head ? 2 : 1;

    // ---- weights -> LDS, transposed (K-major), f16, K zero-padded ----
    for (int idx = tid; idx < H_ * K1_; idx += 128) {
      const int c = idx / K1_, k = idx - c * K1_;
      W1t[c * K1P_ + k] = (_Float16)(k < DIN_ ? w1[k * H_ + c] : 0.f);
    }
    for (int idx = tid; idx < H_ * H_; idx += 128) {
      const int c = idx / H_, k = idx - c * H_;
      W2t[c * K2P_ + k] = (_Float16)w2[k * H_ + c];
    }
    for (int idx = tid; idx < 16 * H_; idx += 128) {
      const int c = idx / H_, k = idx - c * H_;
      W3t[c * K2P_ + k] = (_Float16)(c < ncol3 ? w3[k * ncol3 + c] : 0.f);
    }
    __syncthreads();

    // ---- stage 1: Hs(32x64 per wave) = Xs @ W1 + b1 ----
    {
      v8f acc0[4], acc1[4];
#pragma unroll
      for (int ct = 0; ct < 4; ++ct) { acc0[ct] = zero8(); acc1[ct] = zero8(); }
      const _Float16* xw0 = Xs + (wave * 32) * K1P_;
      const _Float16* xw1 = xw0 + 16 * K1P_;
#pragma unroll
      for (int kk = 0; kk < K1_; kk += 32) {
        // batch all fragment loads of this k-step, then issue 8 WMMAs
        v16h a0 = load_frag_a(xw0, K1P_, kk, lane);
        v16h a1 = load_frag_a(xw1, K1P_, kk, lane);
        v16h bb[4];
#pragma unroll
        for (int ct = 0; ct < 4; ++ct) bb[ct] = load_frag_b(W1t, K1P_, ct * 16, kk, lane);
#pragma unroll
        for (int ct = 0; ct < 4; ++ct) {
          acc0[ct] = wmma_f16(a0, bb[ct], acc0[ct]);
          acc1[ct] = wmma_f16(a1, bb[ct], acc1[ct]);
        }
      }
      const int c0 = lane & 15;
      const int rb = (lane & 16) ? 8 : 0;
#pragma unroll
      for (int ct = 0; ct < 4; ++ct) {
        const int col = ct * 16 + c0;
        const float bias = b1[col];
#pragma unroll
        for (int r = 0; r < 8; ++r) {
          Hs[(wave * 32 + rb + r) * H_ + col]      = acc0[ct][r] + bias;
          Hs[(wave * 32 + 16 + rb + r) * H_ + col] = acc1[ct][r] + bias;
        }
      }
    }
    __syncthreads();

    // ---- LN1 + ReLU -> X2 (one row per thread, all 128 threads) ----
    {
      const float* hr = Hs + tid * H_;
      float mu = 0.f;
      for (int c = 0; c < H_; ++c) mu += hr[c];
      mu *= (1.f / H_);
      float var = 0.f;
      for (int c = 0; c < H_; ++c) { float d = hr[c] - mu; var += d * d; }
      var *= (1.f / H_);
      const float rs = rsqrtf(var + 1e-5f);
      for (int c = 0; c < H_; ++c) {
        float y = (hr[c] - mu) * rs * g1[c] + be1[c];
        X2[tid * K2P_ + c] = (_Float16)fmaxf(y, 0.f);
      }
    }
    __syncthreads();

    // ---- stage 2: Hs = X2 @ W2 + b2 ----
    {
      v8f acc0[4], acc1[4];
#pragma unroll
      for (int ct = 0; ct < 4; ++ct) { acc0[ct] = zero8(); acc1[ct] = zero8(); }
      const _Float16* xw0 = X2 + (wave * 32) * K2P_;
      const _Float16* xw1 = xw0 + 16 * K2P_;
#pragma unroll
      for (int kk = 0; kk < 64; kk += 32) {
        v16h a0 = load_frag_a(xw0, K2P_, kk, lane);
        v16h a1 = load_frag_a(xw1, K2P_, kk, lane);
        v16h bb[4];
#pragma unroll
        for (int ct = 0; ct < 4; ++ct) bb[ct] = load_frag_b(W2t, K2P_, ct * 16, kk, lane);
#pragma unroll
        for (int ct = 0; ct < 4; ++ct) {
          acc0[ct] = wmma_f16(a0, bb[ct], acc0[ct]);
          acc1[ct] = wmma_f16(a1, bb[ct], acc1[ct]);
        }
      }
      const int c0 = lane & 15;
      const int rb = (lane & 16) ? 8 : 0;
#pragma unroll
      for (int ct = 0; ct < 4; ++ct) {
        const int col = ct * 16 + c0;
        const float bias = b2[col];
#pragma unroll
        for (int r = 0; r < 8; ++r) {
          Hs[(wave * 32 + rb + r) * H_ + col]      = acc0[ct][r] + bias;
          Hs[(wave * 32 + 16 + rb + r) * H_ + col] = acc1[ct][r] + bias;
        }
      }
    }
    __syncthreads();

    // ---- LN2 + ReLU -> X2 ----
    {
      const float* hr = Hs + tid * H_;
      float mu = 0.f;
      for (int c = 0; c < H_; ++c) mu += hr[c];
      mu *= (1.f / H_);
      float var = 0.f;
      for (int c = 0; c < H_; ++c) { float d = hr[c] - mu; var += d * d; }
      var *= (1.f / H_);
      const float rs = rsqrtf(var + 1e-5f);
      for (int c = 0; c < H_; ++c) {
        float y = (hr[c] - mu) * rs * g2[c] + be2[c];
        X2[tid * K2P_ + c] = (_Float16)fmaxf(y, 0.f);
      }
    }
    __syncthreads();

    // ---- stage 3: out = X2 @ W3 + b3 (single 16-col tile; cols 0..ncol3-1 live) ----
    {
      v8f acc0 = zero8(), acc1 = zero8();
      const _Float16* xw0 = X2 + (wave * 32) * K2P_;
      const _Float16* xw1 = xw0 + 16 * K2P_;
#pragma unroll
      for (int kk = 0; kk < 64; kk += 32) {
        v16h a0 = load_frag_a(xw0, K2P_, kk, lane);
        v16h a1 = load_frag_a(xw1, K2P_, kk, lane);
        v16h bb = load_frag_b(W3t, K2P_, 0, kk, lane);
        acc0 = wmma_f16(a0, bb, acc0);
        acc1 = wmma_f16(a1, bb, acc1);
      }
      const int col = lane & 15;
      const int rb  = (lane & 16) ? 8 : 0;
      if (head == 0) {
        if (col == 0) {
          const float bias = b3[0];
#pragma unroll
          for (int r = 0; r < 8; ++r) {
            const int na = n0 + wave * 32 + rb + r;
            const int nb = na + 16;
            logits[(size_t)b * N_ + na] = acc0[r] + bias + mask[(size_t)b * N_ + na];
            logits[(size_t)b * N_ + nb] = acc1[r] + bias + mask[(size_t)b * N_ + nb];
          }
        }
      } else {
        if (col < 2) {
          const float bias = b3[col];
#pragma unroll
          for (int r = 0; r < 8; ++r) {
            const int na = n0 + wave * 32 + rb + r;
            const int nb = na + 16;
            offs[((size_t)b * N_ + na) * 2 + col] = acc0[r] + bias;
            offs[((size_t)b * N_ + nb) * 2 + col] = acc1[r] + bias;
          }
        }
      }
    }
    __syncthreads();  // weight buffers reused by next head
  }
}

// ---------------------------------------------------------------------------
// Kernel 2: per-batch softmax stats + BCE partial + smooth-L1 partial.
// grid = B, block = 256.
// ---------------------------------------------------------------------------
__global__ __launch_bounds__(256) void stats_kernel(
    const float* __restrict__ logits, const float* __restrict__ cgt,
    const float* __restrict__ offgt, const float* __restrict__ offs,
    float* __restrict__ bmax, float* __restrict__ bden,
    float* __restrict__ bce_part, float* __restrict__ sl1_part)
{
  __shared__ float red[256];
  __shared__ int   redi[256];
  const int b = blockIdx.x, tid = threadIdx.x;
  const float* lg = logits + (size_t)b * N_;
  const float* gt = cgt + (size_t)b * N_;

  // max
  float m = -3.4e38f;
  for (int n = tid; n < N_; n += 256) m = fmaxf(m, lg[n]);
  red[tid] = m; __syncthreads();
  for (int s = 128; s > 0; s >>= 1) {
    if (tid < s) red[tid] = fmaxf(red[tid], red[tid + s]);
    __syncthreads();
  }
  m = red[0]; __syncthreads();

  // denom
  float sum = 0.f;
  for (int n = tid; n < N_; n += 256) sum += expf(lg[n] - m);
  red[tid] = sum; __syncthreads();
  for (int s = 128; s > 0; s >>= 1) {
    if (tid < s) red[tid] += red[tid + s];
    __syncthreads();
  }
  const float den = red[0]; __syncthreads();
  if (tid == 0) { bmax[b] = m; bden[b] = den; }

  // BCE partial: sum of gt*max(log p,-100) + (1-gt)*max(log1p(-p),-100)
  const float lden = logf(den);
  float acc = 0.f;
  for (int n = tid; n < N_; n += 256) {
    const float l = lg[n] - m;
    const float p = expf(l) / den;
    const float lp  = fmaxf(l - lden, -100.f);
    const float l1p = fmaxf(log1pf(-p), -100.f);
    const float g = gt[n];
    acc += g * lp + (1.f - g) * l1p;
  }
  red[tid] = acc; __syncthreads();
  for (int s = 128; s > 0; s >>= 1) {
    if (tid < s) red[tid] += red[tid + s];
    __syncthreads();
  }
  if (tid == 0) bce_part[b] = red[0];
  __syncthreads();

  // gt index (one-hot argmax == first index where gt > 0.5)
  int gi = N_;
  for (int n = tid; n < N_; n += 256)
    if (gt[n] > 0.5f && n < gi) gi = n;
  redi[tid] = gi; __syncthreads();
  for (int s = 128; s > 0; s >>= 1) {
    if (tid < s) redi[tid] = min(redi[tid], redi[tid + s]);
    __syncthreads();
  }
  if (tid == 0) {
    const int g = redi[0];
    float sl = 0.f;
#pragma unroll
    for (int c = 0; c < 2; ++c) {
      const float d  = offs[((size_t)b * N_ + g) * 2 + c] - offgt[b * 2 + c];
      const float ad = fabsf(d);
      sl += (ad < 1.f) ? 0.5f * d * d : ad - 0.5f;
    }
    sl1_part[b] = sl;
  }
}

// ---------------------------------------------------------------------------
// Kernel 3: per-batch top-50 (iterative argmax, ties -> lower index) + greedy
// NMS exactly in reference order. grid = B, block = 256.
// ---------------------------------------------------------------------------
__global__ __launch_bounds__(256) void topk_nms_kernel(
    const float* __restrict__ logits, const float* __restrict__ bmax, const float* __restrict__ bden,
    const float* __restrict__ cand, const float* __restrict__ offs,
    float* __restrict__ out)
{
  __shared__ float pb[N_];
  __shared__ float rv[256];
  __shared__ int   ri[256];
  __shared__ int   topi[M_];
  __shared__ float tx[M_], ty[M_];

  const int b = blockIdx.x, tid = threadIdx.x;
  const float m = bmax[b], den = bden[b];
  const float* lg = logits + (size_t)b * N_;
  for (int n = tid; n < N_; n += 256) pb[n] = expf(lg[n] - m) / den;
  __syncthreads();

  for (int it = 0; it < M_; ++it) {
    float bv = -3.4e38f; int bi = N_;
    for (int n = tid; n < N_; n += 256) {
      const float v = pb[n];
      if (v > bv) { bv = v; bi = n; }
    }
    rv[tid] = bv; ri[tid] = bi; __syncthreads();
    for (int s = 128; s > 0; s >>= 1) {
      if (tid < s) {
        const float vo = rv[tid + s]; const int io = ri[tid + s];
        if (vo > rv[tid] || (vo == rv[tid] && io < ri[tid])) { rv[tid] = vo; ri[tid] = io; }
      }
      __syncthreads();
    }
    if (tid == 0) { topi[it] = ri[0]; pb[ri[0]] = -3.4e38f; }
    __syncthreads();
  }

  if (tid == 0) {
    for (int i = 0; i < M_; ++i) {
      const size_t base = ((size_t)b * N_ + topi[i]) * 2;
      tx[i] = cand[base + 0] + offs[base + 0];
      ty[i] = cand[base + 1] + offs[base + 1];
    }
    float sx[6], sy[6];
#pragma unroll
    for (int j = 0; j < 6; ++j) { sx[j] = tx[j]; sy[j] = ty[j]; }
    int cnt = 1;
    for (int i = 1; i < M_; ++i) {
      bool close = false;
      for (int j = 0; j < cnt; ++j) {
        const float dx = sx[j] - tx[i], dy = sy[j] - ty[i];
        if (dx * dx + dy * dy < 2.0f) close = true;
      }
      if (!close && cnt < 6) { sx[cnt] = tx[i]; sy[cnt] = ty[i]; ++cnt; }
    }
#pragma unroll
    for (int j = 0; j < 6; ++j) {
      out[((size_t)b * 6 + j) * 2 + 0] = sx[j];
      out[((size_t)b * 6 + j) * 2 + 1] = sy[j];
    }
  }
}

// ---------------------------------------------------------------------------
// Kernel 4: loss = -sum(bce)/(B*N) + sum(sl1)/(B*2). single block of 256.
// ---------------------------------------------------------------------------
__global__ __launch_bounds__(256) void loss_kernel(
    const float* __restrict__ bce_part, const float* __restrict__ sl1_part,
    float* __restrict__ out)
{
  __shared__ float r1[256], r2[256];
  const int tid = threadIdx.x;
  r1[tid] = bce_part[tid];
  r2[tid] = sl1_part[tid];
  __syncthreads();
  for (int s = 128; s > 0; s >>= 1) {
    if (tid < s) { r1[tid] += r1[tid + s]; r2[tid] += r2[tid + s]; }
    __syncthreads();
  }
  if (tid == 0)
    out[B_ * 6 * 2] = -r1[0] / ((float)B_ * (float)N_) + r2[0] / ((float)B_ * 2.f);
}

// ---------------------------------------------------------------------------
extern "C" void kernel_launch(void* const* d_in, const int* in_sizes, int n_in,
                              void* d_out, int out_size, void* d_ws, size_t ws_size,
                              hipStream_t stream) {
  // setup_inputs() dict order:
  const float* feat = (const float*)d_in[0];
  const float* cand = (const float*)d_in[1];
  const float* mask = (const float*)d_in[2];
  const float* cgt  = (const float*)d_in[3];
  const float* ogt  = (const float*)d_in[4];
  const float* pw1  = (const float*)d_in[5];
  const float* pb1  = (const float*)d_in[6];
  const float* pg1  = (const float*)d_in[7];
  const float* pbe1 = (const float*)d_in[8];
  const float* pw2  = (const float*)d_in[9];
  const float* pb2  = (const float*)d_in[10];
  const float* pg2  = (const float*)d_in[11];
  const float* pbe2 = (const float*)d_in[12];
  const float* pw3  = (const float*)d_in[13];
  const float* pb3  = (const float*)d_in[14];
  const float* ow1  = (const float*)d_in[15];
  const float* ob1  = (const float*)d_in[16];
  const float* og1  = (const float*)d_in[17];
  const float* obe1 = (const float*)d_in[18];
  const float* ow2  = (const float*)d_in[19];
  const float* og2  = (const float*)d_in[20];  // NOTE: o head inserts g2 before b2
  const float* ob2  = (const float*)d_in[21];
  const float* obe2 = (const float*)d_in[22];
  const float* ow3  = (const float*)d_in[23];
  const float* ob3  = (const float*)d_in[24];

  float* out = (float*)d_out;

  // workspace layout (floats)
  const size_t BN = (size_t)B_ * N_;
  float* logits   = (float*)d_ws;
  float* offs     = logits + BN;
  float* bmax     = offs + 2 * BN;
  float* bden     = bmax + B_;
  float* bce_part = bden + B_;
  float* sl1_part = bce_part + B_;

  const size_t smem = (size_t)(ROWS_ * K1P_ * 2   // Xs
                             + 64 * K1P_ * 2      // W1t
                             + 64 * K2P_ * 2      // W2t
                             + 16 * K2P_ * 2      // W3t
                             + ROWS_ * K2P_ * 2   // X2
                             + ROWS_ * 64 * 4);   // Hs

  dim3 g1(N_ / ROWS_, B_);
  mlp_kernel<<<g1, 128, smem, stream>>>(
      feat, cand, mask,
      pw1, pb1, pg1, pbe1, pw2, pb2, pg2, pbe2, pw3, pb3,
      ow1, ob1, og1, obe1, ow2, ob2, og2, obe2, ow3, ob3,
      logits, offs);

  stats_kernel<<<B_, 256, 0, stream>>>(logits, cgt, ogt, offs,
                                       bmax, bden, bce_part, sl1_part);

  topk_nms_kernel<<<B_, 256, 0, stream>>>(logits, bmax, bden, cand, offs, out);

  loss_kernel<<<1, 256, 0, stream>>>(bce_part, sl1_part, out);
}
